// PALMAttention_2259152798058
// MI455X (gfx1250) — matrix operand order repaired and checked by reading, not compiled, not run on hardware
//
#include <hip/hip_runtime.h>

// ---------------- problem constants ----------------
#define BDIM   2
#define SDIM   2048
#define HIDD   2048
#define NHEAD  32
#define NKVH   8
#define HD     64
#define GQ     4
#define KVHID  512
#define ROWS   (BDIM * SDIM)          // 4096
#define QSCALE 0.125f                 // 1/sqrt(64)
#define LNEPS  1e-12f

typedef unsigned short u16;
typedef unsigned int   u32;
typedef __attribute__((ext_vector_type(16))) __bf16          v16bf;
typedef __attribute__((ext_vector_type(16))) unsigned short  v16u;
typedef __attribute__((ext_vector_type(8)))  float           v8f;

__device__ __forceinline__ u16 f2bf(float f) {
    u32 u = __float_as_uint(f);
    u32 r = u + 0x7fffu + ((u >> 16) & 1u);   // round-to-nearest-even
    return (u16)(r >> 16);
}

__device__ __forceinline__ v8f v8f_zero() {
    v8f z = {0.f, 0.f, 0.f, 0.f, 0.f, 0.f, 0.f, 0.f};
    return z;
}

__device__ __forceinline__ v16u frag_from_2x16B(const u16* p0, const u16* p1) {
    v16u f;
    ((uint4*)&f)[0] = *(const uint4*)p0;
    ((uint4*)&f)[1] = *(const uint4*)p1;
    return f;
}

// ---------------- fp32 -> bf16 convert (vectorized) ----------------
__global__ __launch_bounds__(256) void cvt_f32_bf16(const float* __restrict__ src,
                                                    u16* __restrict__ dst, int n) {
    int i = blockIdx.x * blockDim.x + threadIdx.x;
    int stride = gridDim.x * blockDim.x;
    const float4* s4 = (const float4*)src;
    uint2* d2 = (uint2*)dst;
    for (int j = i; j < (n >> 2); j += stride) {
        float4 v = s4[j];
        uint2 o;
        o.x = (u32)f2bf(v.x) | ((u32)f2bf(v.y) << 16);
        o.y = (u32)f2bf(v.z) | ((u32)f2bf(v.w) << 16);
        d2[j] = o;
    }
}

// A-tile store: place uint4 quartets in "fragment order":
// dest uints 0..7  = k {0..7,16..23}  (half 0), dest uints 8..15 = k {8..15,24..31} (half 1)
// aligned quartet permutation: uc 0->0, 4->8, 8->4, 12->12  (swap bits 2 and 3)
__device__ __forceinline__ void stageA_store(u16 (*As)[40], int c, uint4 v) {
    int row = c >> 2;
    int uc  = (c & 3) * 4;
    int duc = ((uc & 4) << 1) | ((uc & 8) >> 1);
    *(uint4*)&As[row][duc * 2] = v;
}

// B-tile store transposed: Bt[n][k] (row stride 40 u16 = 80B)
__device__ __forceinline__ void stageB_store(u16 (*Bt)[40], int c, uint4 v) {
    int krow = c >> 4;
    int nc8  = (c & 15) * 8;
    u32 w[4] = {v.x, v.y, v.z, v.w};
#pragma unroll
    for (int j = 0; j < 8; ++j)
        Bt[nc8 + j][krow] = (u16)(w[j >> 1] >> (16 * (j & 1)));
}

// ---------------- bf16 WMMA GEMM: C = A[MxK] @ B[KxN] (+bias)(+resid) ----------------
// 256 threads (8 waves), block tile 128x128, wave tile 32x64, K step 32, LDS double-buffered.
__global__ __launch_bounds__(256) void gemm_bf16_wmma(
    const u16* __restrict__ A, const u16* __restrict__ Bw,
    const float* __restrict__ bias, const float* __restrict__ resid,
    float* __restrict__ Cf, u16* __restrict__ Cb,
    int M, int N, int K)
{
    __shared__ __align__(16) u16 As2[2][128][40];   // fragment-ordered A tiles
    __shared__ __align__(16) u16 Bt[2][128][40];    // transposed B tiles [n][k]

    const int tid  = threadIdx.x;
    const int lane = tid & 31;
    const int wid  = tid >> 5;
    const int wm   = wid >> 1;
    const int wn   = wid & 1;
    const int half = lane >> 4;
    const int l16  = lane & 15;
    const int m0   = blockIdx.y * 128;
    const int n0   = blockIdx.x * 128;
    const int Ku   = K >> 1;
    const int Nu   = N >> 1;

    v8f acc[2][4];
#pragma unroll
    for (int mi = 0; mi < 2; ++mi)
#pragma unroll
        for (int ni = 0; ni < 4; ++ni) acc[mi][ni] = v8f_zero();

    const u32* Aui = (const u32*)A;
    const u32* Bui = (const u32*)Bw;
    const int  T   = K >> 5;

    // ---- stage tile 0 ----
#pragma unroll
    for (int it = 0; it < 2; ++it) {
        int c = tid + 256 * it;
        uint4 va = *(const uint4*)(Aui + (size_t)(m0 + (c >> 2)) * Ku + (c & 3) * 4);
        stageA_store(As2[0], c, va);
        uint4 vb = *(const uint4*)(Bui + (size_t)(c >> 4) * Nu + ((n0 + (c & 15) * 8) >> 1));
        stageB_store(Bt[0], c, vb);
    }
    __syncthreads();

    for (int t = 0; t < T; ++t) {
        const int cur  = t & 1;
        const int nxt  = cur ^ 1;
        const bool more = (t + 1 < T);
        uint4 va[2], vb[2];
        if (more) {
            const int ktn = (t + 1) << 5;
#pragma unroll
            for (int it = 0; it < 2; ++it) {
                int c = tid + 256 * it;
                va[it] = *(const uint4*)(Aui + (size_t)(m0 + (c >> 2)) * Ku + (ktn >> 1) + (c & 3) * 4);
                vb[it] = *(const uint4*)(Bui + (size_t)(ktn + (c >> 4)) * Nu + ((n0 + (c & 15) * 8) >> 1));
            }
        }

        // ---- fragments: each = two ds_load_b128 ----
        v16u afr[2], bfr[4];
#pragma unroll
        for (int mi = 0; mi < 2; ++mi) {
            const u16* p = &As2[cur][wm * 32 + mi * 16 + l16][half * 16];
            afr[mi] = frag_from_2x16B(p, p + 8);
        }
#pragma unroll
        for (int ni = 0; ni < 4; ++ni) {
            const u16* p = &Bt[cur][wn * 64 + ni * 16 + l16][half * 8];
            bfr[ni] = frag_from_2x16B(p, p + 16);
        }
#pragma unroll
        for (int mi = 0; mi < 2; ++mi)
#pragma unroll
            for (int ni = 0; ni < 4; ++ni)
                acc[mi][ni] = __builtin_amdgcn_wmma_f32_16x16x32_bf16(
                    false, __builtin_bit_cast(v16bf, afr[mi]),
                    false, __builtin_bit_cast(v16bf, bfr[ni]),
                    (short)0, acc[mi][ni], false, false);

        if (more) {
#pragma unroll
            for (int it = 0; it < 2; ++it) {
                int c = tid + 256 * it;
                stageA_store(As2[nxt], c, va[it]);
                stageB_store(Bt[nxt], c, vb[it]);
            }
        }
        __syncthreads();
    }

    // ---- epilogue ----
#pragma unroll
    for (int mi = 0; mi < 2; ++mi) {
#pragma unroll
        for (int ni = 0; ni < 4; ++ni) {
#pragma unroll
            for (int r = 0; r < 8; ++r) {
                int row = m0 + wm * 32 + mi * 16 + r + 8 * half;
                int col = n0 + wn * 64 + ni * 16 + l16;
                float v = acc[mi][ni][r];
                if (bias)  v += bias[col];
                if (resid) v += resid[(size_t)row * N + col];
                if (Cf) Cf[(size_t)row * N + col] = v;
                if (Cb) Cb[(size_t)row * N + col] = f2bf(v);
            }
        }
    }
}

// ---------------- flash attention (GQA), bf16 WMMA, double-buffered K/V ----------------
// grid: (S/64, NHEAD, B); block: 128 threads (4 waves); wave = 16 query rows.
__global__ __launch_bounds__(128) void attn_wmma(
    const u16* __restrict__ Q, const u16* __restrict__ Kb,
    const u16* __restrict__ Vb, const float* __restrict__ maskp,
    u16* __restrict__ Ctx)
{
    __shared__ __align__(16) u16 Ks[2][32][72];     // [key][d], padded
    __shared__ __align__(16) u16 Vt[2][64][40];     // transposed [d][key], padded
    __shared__ __align__(16) u16 Ps[4][16][32];     // per-wave P tile
    __shared__ float Msk[2][32];

    const int tid  = threadIdx.x;
    const int lane = tid & 31;
    const int wid  = tid >> 5;
    const int half = lane >> 4;
    const int l16  = lane & 15;
    const int qs0  = blockIdx.x * 64;
    const int h    = blockIdx.y;
    const int bb   = blockIdx.z;
    const int kv   = h >> 2;

    // ---- Q A-fragments: two global_load_b128 per fragment ----
    const int qrow = bb * SDIM + qs0 + wid * 16 + l16;
    const u16* qp  = Q + (size_t)qrow * HIDD + h * HD;
    v16u qfr[2];
#pragma unroll
    for (int kk = 0; kk < 2; ++kk) {
        const u16* p = qp + kk * 32 + 8 * half;
        qfr[kk] = frag_from_2x16B(p, p + 16);
    }

    v8f ctx[4];
#pragma unroll
    for (int ni = 0; ni < 4; ++ni) ctx[ni] = v8f_zero();
    float mrow[8], lrow[8];
#pragma unroll
    for (int r = 0; r < 8; ++r) { mrow[r] = -1e30f; lrow[r] = 0.f; }

    const u32* Kui = (const u32*)Kb;
    const u32* Vui = (const u32*)Vb;
    const int  T   = SDIM / 32;

    // ---- stage chunk 0 ----
#pragma unroll
    for (int it = 0; it < 2; ++it) {
        int c   = tid + 128 * it;       // 0..255
        int row = c >> 3;
        int dc  = (c & 7) * 8;
        size_t g = (size_t)(bb * SDIM + row) * (KVHID / 2) + ((kv * HD + dc) >> 1);
        uint4 kvv = *(const uint4*)(Kui + g);
        *(uint4*)&Ks[0][row][dc] = kvv;
        uint4 vvv = *(const uint4*)(Vui + g);
        u32 w[4] = {vvv.x, vvv.y, vvv.z, vvv.w};
#pragma unroll
        for (int j = 0; j < 8; ++j) Vt[0][dc + j][row] = (u16)(w[j >> 1] >> (16 * (j & 1)));
    }
    if (tid < 32) Msk[0][tid] = maskp[bb * SDIM + tid];
    __syncthreads();

    for (int t = 0; t < T; ++t) {
        const int cur  = t & 1;
        const int nxt  = cur ^ 1;
        const bool more = (t + 1 < T);
        uint4 kr[2], vr[2];
        float mr = 0.f;
        if (more) {
            const int key1 = (t + 1) * 32;
#pragma unroll
            for (int it = 0; it < 2; ++it) {
                int c   = tid + 128 * it;
                int row = c >> 3;
                int dc  = (c & 7) * 8;
                size_t g = (size_t)(bb * SDIM + key1 + row) * (KVHID / 2) + ((kv * HD + dc) >> 1);
                kr[it] = *(const uint4*)(Kui + g);
                vr[it] = *(const uint4*)(Vui + g);
            }
            if (tid < 32) mr = maskp[bb * SDIM + key1 + tid];
            if (t + 2 < T) {    // prefetch chunk t+2 into L2 (global_prefetch_b8)
                size_t g2 = (size_t)(bb * SDIM + (t + 2) * 32 + (tid >> 3)) * (KVHID / 2)
                          + ((kv * HD + (tid & 7) * 8) >> 1);
                __builtin_prefetch((const void*)(Kui + g2), 0, 0);
                __builtin_prefetch((const void*)(Vui + g2), 0, 0);
            }
        }

        // ---- scores: Q(16x64) @ K^T(64x32) ----
        v8f sfr[2];
#pragma unroll
        for (int nt = 0; nt < 2; ++nt) {
            v8f s = v8f_zero();
            const int nkey = nt * 16 + l16;
#pragma unroll
            for (int kk = 0; kk < 2; ++kk) {
                const u16* p = &Ks[cur][nkey][kk * 32 + 8 * half];
                v16u bf = frag_from_2x16B(p, p + 16);
                s = __builtin_amdgcn_wmma_f32_16x16x32_bf16(
                    false, __builtin_bit_cast(v16bf, qfr[kk]),
                    false, __builtin_bit_cast(v16bf, bf),
                    (short)0, s, false, false);
            }
            sfr[nt] = s;
        }

        // ---- online softmax ----
        float alpha[8];
#pragma unroll
        for (int r = 0; r < 8; ++r) {
            float s0 = sfr[0][r] * QSCALE + Msk[cur][l16];
            float s1 = sfr[1][r] * QSCALE + Msk[cur][16 + l16];
            float mx = fmaxf(s0, s1);
#pragma unroll
            for (int off = 1; off < 16; off <<= 1) mx = fmaxf(mx, __shfl_xor(mx, off, 32));
            float nm = fmaxf(mrow[r], mx);
            alpha[r] = __expf(mrow[r] - nm);
            float p0 = __expf(s0 - nm);
            float p1 = __expf(s1 - nm);
            sfr[0][r] = p0; sfr[1][r] = p1;
            float rssum = p0 + p1;
#pragma unroll
            for (int off = 1; off < 16; off <<= 1) rssum += __shfl_xor(rssum, off, 32);
            lrow[r] = lrow[r] * alpha[r] + rssum;
            mrow[r] = nm;
        }
        // ---- P to LDS (C layout), rescale ctx ----
#pragma unroll
        for (int r = 0; r < 8; ++r) {
            Ps[wid][r + 8 * half][l16]      = f2bf(sfr[0][r]);
            Ps[wid][r + 8 * half][16 + l16] = f2bf(sfr[1][r]);
        }
#pragma unroll
        for (int ni = 0; ni < 4; ++ni)
#pragma unroll
            for (int r = 0; r < 8; ++r) ctx[ni][r] *= alpha[r];

        // ---- P fragment: two ds_load_b128 ----
        const u16* pp = &Ps[wid][l16][8 * half];
        v16u pfr = frag_from_2x16B(pp, pp + 16);

        // ---- ctx += P(16x32) @ V(32x64): V frags from transposed LDS ----
#pragma unroll
        for (int ni = 0; ni < 4; ++ni) {
            const u16* p = &Vt[cur][ni * 16 + l16][8 * half];
            v16u vf = frag_from_2x16B(p, p + 16);
            ctx[ni] = __builtin_amdgcn_wmma_f32_16x16x32_bf16(
                false, __builtin_bit_cast(v16bf, pfr),
                false, __builtin_bit_cast(v16bf, vf),
                (short)0, ctx[ni], false, false);
        }

        if (more) {
#pragma unroll
            for (int it = 0; it < 2; ++it) {
                int c   = tid + 128 * it;
                int row = c >> 3;
                int dc  = (c & 7) * 8;
                *(uint4*)&Ks[nxt][row][dc] = kr[it];
                u32 w[4] = {vr[it].x, vr[it].y, vr[it].z, vr[it].w};
#pragma unroll
                for (int j = 0; j < 8; ++j) Vt[nxt][dc + j][row] = (u16)(w[j >> 1] >> (16 * (j & 1)));
            }
            if (tid < 32) Msk[nxt][tid] = mr;
        }
        __syncthreads();
    }

    // ---- normalize + store bf16 context ----
#pragma unroll
    for (int ni = 0; ni < 4; ++ni) {
#pragma unroll
        for (int r = 0; r < 8; ++r) {
            float inv = (lrow[r] > 0.f) ? 1.f / lrow[r] : 0.f;
            int srow = qs0 + wid * 16 + r + 8 * half;
            int col  = h * HD + ni * 16 + l16;
            Ctx[(size_t)(bb * SDIM + srow) * HIDD + col] = f2bf(ctx[ni][r] * inv);
        }
    }
}

// ---------------- LayerNorm over last dim (2048), one block per row ----------------
__global__ __launch_bounds__(256) void ln_kernel(const float* __restrict__ X,
                                                 const float* __restrict__ gamma,
                                                 const float* __restrict__ beta,
                                                 float* __restrict__ Y)
{
    __shared__ float rs[8], rss[8];
    const int row = blockIdx.x;
    const float4* x4 = (const float4*)(X + (size_t)row * HIDD);
    float s = 0.f, ss = 0.f;
    for (int i = threadIdx.x; i < HIDD / 4; i += 256) {
        float4 v = x4[i];
        s  += v.x + v.y + v.z + v.w;
        ss += v.x * v.x + v.y * v.y + v.z * v.z + v.w * v.w;
    }
#pragma unroll
    for (int off = 16; off; off >>= 1) { s += __shfl_xor(s, off, 32); ss += __shfl_xor(ss, off, 32); }
    int wid = threadIdx.x >> 5;
    if ((threadIdx.x & 31) == 0) { rs[wid] = s; rss[wid] = ss; }
    __syncthreads();
    if (threadIdx.x == 0) {
        float a = 0.f, c = 0.f;
        for (int i = 0; i < 8; ++i) { a += rs[i]; c += rss[i]; }
        rs[0] = a; rss[0] = c;
    }
    __syncthreads();
    float mean = rs[0] * (1.f / HIDD);
    float var  = rss[0] * (1.f / HIDD) - mean * mean;
    float inv  = rsqrtf(var + LNEPS);
    const float4* g4 = (const float4*)gamma;
    const float4* b4 = (const float4*)beta;
    float4* y4 = (float4*)(Y + (size_t)row * HIDD);
    for (int i = threadIdx.x; i < HIDD / 4; i += 256) {
        float4 v = x4[i], g = g4[i], b = b4[i];
        float4 o;
        o.x = (v.x - mean) * inv * g.x + b.x;
        o.y = (v.y - mean) * inv * g.y + b.y;
        o.z = (v.z - mean) * inv * g.z + b.z;
        o.w = (v.w - mean) * inv * g.w + b.w;
        if (!__builtin_isfinite(o.x)) o.x = 0.f;
        if (!__builtin_isfinite(o.y)) o.y = 0.f;
        if (!__builtin_isfinite(o.z)) o.z = 0.f;
        if (!__builtin_isfinite(o.w)) o.w = 0.f;
        y4[i] = o;
    }
}

// ---------------- launcher ----------------
extern "C" void kernel_launch(void* const* d_in, const int* in_sizes, int n_in,
                              void* d_out, int out_size, void* d_ws, size_t ws_size,
                              hipStream_t stream) {
    const float* hs    = (const float*)d_in[0];
    const float* mask  = (const float*)d_in[1];
    const float* Wq    = (const float*)d_in[2];
    const float* bq    = (const float*)d_in[3];
    const float* Wk    = (const float*)d_in[4];
    const float* bk    = (const float*)d_in[5];
    const float* Wv    = (const float*)d_in[6];
    const float* bv    = (const float*)d_in[7];
    const float* Wd    = (const float*)d_in[8];
    const float* bd    = (const float*)d_in[9];
    const float* gamma = (const float*)d_in[10];
    const float* beta  = (const float*)d_in[11];

    char* ws = (char*)d_ws;
    size_t off = 0;
    auto alloc = [&](size_t bytes) -> void* {
        void* p = ws + off;
        off += (bytes + 255) & ~(size_t)255;
        return p;
    };
    u16*   xb  = (u16*)alloc((size_t)ROWS * HIDD * 2);
    u16*   Wqb = (u16*)alloc((size_t)HIDD * HIDD * 2);
    u16*   Wkb = (u16*)alloc((size_t)HIDD * KVHID * 2);
    u16*   Wvb = (u16*)alloc((size_t)HIDD * KVHID * 2);
    u16*   Wdb = (u16*)alloc((size_t)HIDD * HIDD * 2);
    u16*   qb  = (u16*)alloc((size_t)ROWS * HIDD * 2);
    u16*   kb  = (u16*)alloc((size_t)ROWS * KVHID * 2);
    u16*   vb  = (u16*)alloc((size_t)ROWS * KVHID * 2);
    u16*   cb  = (u16*)alloc((size_t)ROWS * HIDD * 2);
    float* xr  = (float*)alloc((size_t)ROWS * HIDD * 4);

    const int CB = 2048;
    cvt_f32_bf16<<<CB, 256, 0, stream>>>(hs, xb, ROWS * HIDD);
    cvt_f32_bf16<<<CB, 256, 0, stream>>>(Wq, Wqb, HIDD * HIDD);
    cvt_f32_bf16<<<CB, 256, 0, stream>>>(Wk, Wkb, HIDD * KVHID);
    cvt_f32_bf16<<<CB, 256, 0, stream>>>(Wv, Wvb, HIDD * KVHID);
    cvt_f32_bf16<<<CB, 256, 0, stream>>>(Wd, Wdb, HIDD * HIDD);

    gemm_bf16_wmma<<<dim3(HIDD / 128, ROWS / 128), 256, 0, stream>>>(
        xb, Wqb, bq, nullptr, nullptr, qb, ROWS, HIDD, HIDD);
    gemm_bf16_wmma<<<dim3(KVHID / 128, ROWS / 128), 256, 0, stream>>>(
        xb, Wkb, bk, nullptr, nullptr, kb, ROWS, KVHID, HIDD);
    gemm_bf16_wmma<<<dim3(KVHID / 128, ROWS / 128), 256, 0, stream>>>(
        xb, Wvb, bv, nullptr, nullptr, vb, ROWS, KVHID, HIDD);

    attn_wmma<<<dim3(SDIM / 64, NHEAD, BDIM), 128, 0, stream>>>(qb, kb, vb, mask, cb);

    gemm_bf16_wmma<<<dim3(HIDD / 128, ROWS / 128), 256, 0, stream>>>(
        cb, Wdb, bd, hs, xr, nullptr, ROWS, HIDD, HIDD);

    ln_kernel<<<ROWS, 256, 0, stream>>>(xr, gamma, beta, (float*)d_out);
}
